// MultiHeadAttention_54296976556288
// MI455X (gfx1250) — compile-verified, matrix operand order checked by reading, not run on hardware
//
#include <hip/hip_runtime.h>
#include <hip/hip_bf16.h>
#include <math.h>

// ---------------------------------------------------------------------------
// MI455X (gfx1250) fused attention block.
// GEMMs via v_wmma_f32_16x16x32_bf16 (wave32, f32 accum); attention K-chunks
// staged to LDS with the Tensor Data Mover (tensor_load_to_lds + TENSORcnt).
// GEMM inner loops are unrolled-by-2 double-buffered so prefetched fragments
// land directly in their WMMA operand registers (no rotation copies).
// ---------------------------------------------------------------------------

typedef __bf16 bf16_t;
typedef bf16_t v16bf __attribute__((ext_vector_type(16)));
typedef bf16_t v8bf  __attribute__((ext_vector_type(8)));
typedef float  v8f   __attribute__((ext_vector_type(8)));
typedef unsigned int v4u __attribute__((ext_vector_type(4)));
typedef int v8i __attribute__((ext_vector_type(8)));
typedef int v4i __attribute__((ext_vector_type(4)));

#define BATCH   4
#define SEQ     2048
#define DMODEL  1024
#define NHEADS  16
#define HDIM    64
#define NROWS   (BATCH * SEQ)   // 8192

// ---------------------------------------------------------------------------
// Fragment loader for CDNA5 16-bit WMMA operands.
// A (16x32, M x K): lane%16 selects row M, per-lane K = {0..7,16..23}+8*half.
// B (32x16, K x N) when the source is stored [N][K] row-major: identical
// pattern with lane%16 selecting the N row.  Two contiguous 16B loads/lane.
// ---------------------------------------------------------------------------
__device__ __forceinline__ v16bf ld_frag(const bf16_t* base, int ld, int row, int k0) {
  const int lane = threadIdx.x & 31;
  const int l16  = lane & 15;
  const int half = lane >> 4;
  const bf16_t* p = base + (size_t)(row + l16) * (size_t)ld + (k0 + half * 8);
  v8bf lo = *(const v8bf*)p;
  v8bf hi = *(const v8bf*)(p + 16);
  v16bf r;
#pragma unroll
  for (int i = 0; i < 8; ++i) { r[i] = lo[i]; r[i + 8] = hi[i]; }
  return r;
}

__device__ __forceinline__ v8f wmma_bf16(v16bf a, v16bf b, v8f c) {
  return __builtin_amdgcn_wmma_f32_16x16x32_bf16(
      /*neg_a=*/false, a, /*neg_b=*/false, b,
      /*c_mod=*/(short)0, c, /*reuse_a=*/false, /*reuse_b=*/false);
}

// ---------------------------------------------------------------------------
// Tensor Data Mover: 2D tile (tile_h rows x tile_w bf16 elems) from global
// (row stride = stride_elems) into LDS at lds_byte_addr, packed row-major.
// D# built per CDNA5 ISA (group0 + group1; groups 2/3
// zero for a 2D tensor).  Tracked with TENSORcnt.  This toolchain exposes the
// 6-arg builtin: (v4u g0, v8i g1, v4i g2, v4i g3, v8i g4, i32 cpol).
// ---------------------------------------------------------------------------
__device__ __forceinline__ void tdm_load_tile_2d(unsigned lds_byte_addr,
                                                 const bf16_t* gptr,
                                                 unsigned tile_w, unsigned tile_h,
                                                 unsigned stride_elems) {
  unsigned long long ga = (unsigned long long)(uintptr_t)gptr;
  v4u g0;
  g0[0] = 1u;                                            // count=1, user desc
  g0[1] = lds_byte_addr;                                 // lds_addr
  g0[2] = (unsigned)(ga & 0xffffffffu);                  // global_addr[31:0]
  g0[3] = (unsigned)((ga >> 32) & 0x01ffffffu)           // global_addr[56:32]
          | (2u << 30);                                  // type = 2 (image)
  v8i g1;
  g1[0] = (int)(1u << 16);                               // data_size=1 -> 2B
  g1[1] = (int)((stride_elems & 0xffffu) << 16);         // tensor_dim0[15:0]
  g1[2] = (int)(((stride_elems >> 16) & 0xffffu)         // tensor_dim0[31:16]
          | (0xffffu << 16));                            // tensor_dim1[15:0]
  g1[3] = (int)(0x7fffu | (tile_w << 16));               // dim1 hi | tile_dim0
  g1[4] = (int)(tile_h & 0xffffu);                       // tile_dim1 (tile_dim2=0)
  g1[5] = (int)stride_elems;                             // tensor_dim0_stride lo
  g1[6] = 0;                                             // stride hi | dim1_stride lo
  g1[7] = 0;                                             // dim1_stride hi
  v4i gz4 = {0, 0, 0, 0};
  v8i gz8 = {0, 0, 0, 0, 0, 0, 0, 0};
  __builtin_amdgcn_tensor_load_to_lds(g0, g1, gz4, gz4, gz8, 0);
}

// ---------------------------------------------------------------------------
// Stage 0a: x (f32) -> xb (bf16), vectorized.
// ---------------------------------------------------------------------------
__global__ void k_cvt_x(const float* __restrict__ x, bf16_t* __restrict__ xb, int n) {
  int i = (blockIdx.x * blockDim.x + threadIdx.x) * 4;
  if (i + 3 < n) {
    float4 v = *(const float4*)(x + i);
    xb[i + 0] = (bf16_t)v.x;
    xb[i + 1] = (bf16_t)v.y;
    xb[i + 2] = (bf16_t)v.z;
    xb[i + 3] = (bf16_t)v.w;
  }
}

// ---------------------------------------------------------------------------
// Stage 0b: weight [K][N] f32 -> wT [N][K] bf16 (LDS-tiled transpose).
// ---------------------------------------------------------------------------
__global__ void k_cvt_wT(const float* __restrict__ w, bf16_t* __restrict__ wT) {
  __shared__ float tile[32][33];
  const int n0 = blockIdx.x * 32;
  const int k0 = blockIdx.y * 32;
  const int tx = threadIdx.x & 31;
  const int ty = threadIdx.x >> 5;   // 0..7
#pragma unroll
  for (int j = ty; j < 32; j += 8)
    tile[j][tx] = w[(size_t)(k0 + j) * DMODEL + (n0 + tx)];
  __syncthreads();
#pragma unroll
  for (int j = ty; j < 32; j += 8)
    wT[(size_t)(n0 + j) * DMODEL + (k0 + tx)] = (bf16_t)tile[tx][j];
}

// ---------------------------------------------------------------------------
// Stage 1: Q/K/V = xb @ W.  Block = 4 waves, 128x64 C tile; each wave owns a
// 32x64 strip (8 WMMA / 32-wide k-step).  Unroll-by-2 double buffering: the
// two phases use disjoint fragment registers, so prefetch loads write their
// WMMA operands directly.
// ---------------------------------------------------------------------------
__global__ void __launch_bounds__(128) k_qkv(
    const bf16_t* __restrict__ xb,
    const bf16_t* __restrict__ wqT, const bf16_t* __restrict__ wkT,
    const bf16_t* __restrict__ wvT,
    bf16_t* __restrict__ Q, bf16_t* __restrict__ K, bf16_t* __restrict__ V) {
  const bf16_t* wT;
  bf16_t* out;
  if (blockIdx.z == 0)      { wT = wqT; out = Q; }
  else if (blockIdx.z == 1) { wT = wkT; out = K; }
  else                      { wT = wvT; out = V; }

  const int wave = threadIdx.x >> 5;               // 0..3
  const int m0 = blockIdx.y * 128 + wave * 32;
  const int n0 = blockIdx.x * 64;

  v8f acc[8] = {};

  // Phase-0 buffer (k), phase-1 buffer (k+32).
  v16bf a0 = ld_frag(xb, DMODEL, m0,      0);
  v16bf a1 = ld_frag(xb, DMODEL, m0 + 16, 0);
  v16bf b0[4];
#pragma unroll
  for (int t = 0; t < 4; ++t) b0[t] = ld_frag(wT, DMODEL, n0 + t * 16, 0);

  for (int k = 0; k < DMODEL; k += 64) {
    // Prefetch phase-1 operands (k+32 always in range: k <= 960).
    v16bf c0 = ld_frag(xb, DMODEL, m0,      k + 32);
    v16bf c1 = ld_frag(xb, DMODEL, m0 + 16, k + 32);
    v16bf b1[4];
#pragma unroll
    for (int t = 0; t < 4; ++t) b1[t] = ld_frag(wT, DMODEL, n0 + t * 16, k + 32);

    // Compute phase 0.
#pragma unroll
    for (int t = 0; t < 4; ++t) {
      acc[t]     = wmma_bf16(a0, b0[t], acc[t]);
      acc[4 + t] = wmma_bf16(a1, b0[t], acc[4 + t]);
    }

    // Prefetch next iteration's phase-0 operands.
    const int k2 = (k + 64 < DMODEL) ? (k + 64) : 0;
    a0 = ld_frag(xb, DMODEL, m0,      k2);
    a1 = ld_frag(xb, DMODEL, m0 + 16, k2);
#pragma unroll
    for (int t = 0; t < 4; ++t) b0[t] = ld_frag(wT, DMODEL, n0 + t * 16, k2);

    // Compute phase 1.
#pragma unroll
    for (int t = 0; t < 4; ++t) {
      acc[t]     = wmma_bf16(c0, b1[t], acc[t]);
      acc[4 + t] = wmma_bf16(c1, b1[t], acc[4 + t]);
    }
  }

  const int lane = threadIdx.x & 31, l16 = lane & 15, half = lane >> 4;
#pragma unroll
  for (int mi = 0; mi < 2; ++mi)
#pragma unroll
    for (int t = 0; t < 4; ++t)
#pragma unroll
      for (int j = 0; j < 8; ++j) {
        int row = m0 + mi * 16 + half * 8 + j;
        int col = n0 + t * 16 + l16;
        out[(size_t)row * DMODEL + col] = (bf16_t)acc[mi * 4 + t][j];
      }
}

// ---------------------------------------------------------------------------
// Stage 2: flash attention.  Block = 4 waves; each wave owns 16 Q rows of one
// (b,h).  Per 32-wide KV chunk: wave 0 DMAs the K chunk [32x64] into LDS via
// the Tensor Data Mover; V chunk is cooperatively transposed into LDS; scores
// go through per-wave LDS for the online softmax; P@V accumulates via WMMA.
// ---------------------------------------------------------------------------
__global__ void __launch_bounds__(128) k_attn(
    const bf16_t* __restrict__ Q, const bf16_t* __restrict__ K,
    const bf16_t* __restrict__ V, bf16_t* __restrict__ O) {
  __shared__ __align__(16) bf16_t Kc[32 * HDIM];       // [kv_local][d] via TDM
  __shared__ __align__(16) bf16_t Vt[HDIM * 32];       // [d][kv_local]
  __shared__ __align__(16) float  Ssh[4][16 * 32];     // per-wave scores
  __shared__ __align__(16) bf16_t Psh[4][16 * 32];     // per-wave probs
  __shared__ float Fsh[4][16];                          // per-wave row factors

  const int wave = threadIdx.x >> 5;
  const int lane = threadIdx.x & 31, l16 = lane & 15, half = lane >> 4;
  const int b = blockIdx.y >> 4;
  const int h = blockIdx.y & 15;

  const size_t base = ((size_t)b * SEQ) * DMODEL + (size_t)h * HDIM;
  const bf16_t* Qb = Q + base;
  const bf16_t* Kb = K + base;
  const bf16_t* Vb = V + base;
  bf16_t*       Ob = O + base;

  const int q0 = blockIdx.x * 64 + wave * 16;
  const unsigned kc_lds = (unsigned)(uintptr_t)&Kc[0];

  // Q fragments live in registers for the whole KV loop (Dh=64 -> 2 K-steps).
  v16bf aq0 = ld_frag(Qb, DMODEL, q0, 0);
  v16bf aq1 = ld_frag(Qb, DMODEL, q0, 32);

  v8f o0 = {}, o1 = {}, o2 = {}, o3 = {};
  float m_r = -1e30f, l_r = 0.f;
  const float scale = 0.125f;   // 1/sqrt(64)

  for (int kv = 0; kv < SEQ; kv += 32) {
    __syncthreads();  // Kc/Vt from previous chunk fully consumed

    // K chunk [32 x 64] -> LDS via Tensor Data Mover (wave 0 issues).
    if (wave == 0)
      tdm_load_tile_2d(kc_lds, Kb + (size_t)kv * DMODEL, HDIM, 32, DMODEL);

    // Cooperative transposed load of the V chunk: Vt[d][kv_local].
    {
      const int r  = threadIdx.x >> 2;          // kv row 0..31
      const int c0 = (threadIdx.x & 3) * 16;    // d column group
      const bf16_t* vp = Vb + (size_t)(kv + r) * DMODEL + c0;
      v8bf v0 = *(const v8bf*)vp;
      v8bf v1 = *(const v8bf*)(vp + 8);
#pragma unroll
      for (int i = 0; i < 8; ++i) {
        Vt[(c0 + i) * 32 + r]     = v0[i];
        Vt[(c0 + 8 + i) * 32 + r] = v1[i];
      }
    }
    if (wave == 0) __builtin_amdgcn_s_wait_tensorcnt(0);
    __syncthreads();  // Kc (TDM) + Vt ready for all waves

    // S = Q @ K^T from the LDS K chunk (two 16x16 tiles along KV).
    v8f s0 = {}, s1 = {};
    {
      v16bf b00 = ld_frag(Kc, HDIM, 0,  0);
      v16bf b01 = ld_frag(Kc, HDIM, 0,  32);
      v16bf b10 = ld_frag(Kc, HDIM, 16, 0);
      v16bf b11 = ld_frag(Kc, HDIM, 16, 32);
      s0 = wmma_bf16(aq0, b00, s0);
      s0 = wmma_bf16(aq1, b01, s0);
      s1 = wmma_bf16(aq0, b10, s1);
      s1 = wmma_bf16(aq1, b11, s1);
    }
#pragma unroll
    for (int j = 0; j < 8; ++j) {
      Ssh[wave][(half * 8 + j) * 32 + l16]      = s0[j];
      Ssh[wave][(half * 8 + j) * 32 + 16 + l16] = s1[j];
    }
    __syncthreads();

    // Online softmax: lanes 0..15 each own one Q row of this wave.
    if (lane < 16) {
      float mx = m_r;
#pragma unroll
      for (int i = 0; i < 32; ++i)
        mx = fmaxf(mx, Ssh[wave][lane * 32 + i] * scale);
      float alpha = __expf(m_r - mx);
      float sum = 0.f;
#pragma unroll
      for (int i = 0; i < 32; ++i) {
        float p = __expf(Ssh[wave][lane * 32 + i] * scale - mx);
        Psh[wave][lane * 32 + i] = (bf16_t)p;
        sum += p;
      }
      m_r = mx;
      l_r = l_r * alpha + sum;
      Fsh[wave][lane] = alpha;
    }
    __syncthreads();

    // Rescale running O, then O += P @ V_chunk.
    v16bf ap  = ld_frag(Psh[wave], 32, 0, 0);
    v16bf bv0 = ld_frag(Vt, 32, 0,  0);
    v16bf bv1 = ld_frag(Vt, 32, 16, 0);
    v16bf bv2 = ld_frag(Vt, 32, 32, 0);
    v16bf bv3 = ld_frag(Vt, 32, 48, 0);
#pragma unroll
    for (int j = 0; j < 8; ++j) {
      float f = Fsh[wave][half * 8 + j];
      o0[j] *= f; o1[j] *= f; o2[j] *= f; o3[j] *= f;
    }
    o0 = wmma_bf16(ap, bv0, o0);
    o1 = wmma_bf16(ap, bv1, o1);
    o2 = wmma_bf16(ap, bv2, o2);
    o3 = wmma_bf16(ap, bv3, o3);
  }

  // 1/l normalization and store O (bf16, heads concatenated in model layout).
  if (lane < 16) Fsh[wave][lane] = (l_r > 0.f) ? (1.f / l_r) : 0.f;
  __syncthreads();
  v8f* oacc[4] = { &o0, &o1, &o2, &o3 };
#pragma unroll
  for (int t = 0; t < 4; ++t)
#pragma unroll
    for (int j = 0; j < 8; ++j) {
      int row = q0 + half * 8 + j;
      int col = t * 16 + l16;
      Ob[(size_t)row * DMODEL + col] = (bf16_t)((*oacc[t])[j] * Fsh[wave][half * 8 + j]);
    }
}

// ---------------------------------------------------------------------------
// Stage 3: y = O @ Wp + b + x, then LayerNorm per row.  Block = 8 waves
// computing a full 16x1024 row strip so the LN reduction stays in LDS
// (ds_add_f32 atomics over 16 row accumulators).  Unroll-by-2 A ping-pong.
// ---------------------------------------------------------------------------
__global__ void __launch_bounds__(256) k_proj_ln(
    const bf16_t* __restrict__ Ob, const bf16_t* __restrict__ wpT,
    const float* __restrict__ b_proj, const float* __restrict__ x,
    const float* __restrict__ gamma, const float* __restrict__ beta,
    float* __restrict__ out) {
  __shared__ float Sum[16], Sq[16];

  const int wave = threadIdx.x >> 5;               // 0..7
  const int lane = threadIdx.x & 31, l16 = lane & 15, half = lane >> 4;
  const int r0 = blockIdx.x * 16;
  const int n0 = wave * 128;

  v8f acc[8] = {};
  v16bf a0 = ld_frag(Ob, DMODEL, r0, 0);
  for (int k = 0; k < DMODEL; k += 64) {
    v16bf a1 = ld_frag(Ob, DMODEL, r0, k + 32);
#pragma unroll
    for (int t = 0; t < 8; ++t) {
      v16bf b = ld_frag(wpT, DMODEL, n0 + t * 16, k);
      acc[t] = wmma_bf16(a0, b, acc[t]);
    }
    const int k2 = (k + 64 < DMODEL) ? (k + 64) : 0;
    a0 = ld_frag(Ob, DMODEL, r0, k2);
#pragma unroll
    for (int t = 0; t < 8; ++t) {
      v16bf b = ld_frag(wpT, DMODEL, n0 + t * 16, k + 32);
      acc[t] = wmma_bf16(a1, b, acc[t]);
    }
  }

  if (threadIdx.x < 16) { Sum[threadIdx.x] = 0.f; Sq[threadIdx.x] = 0.f; }
  __syncthreads();

  float psum[8] = {}, psq[8] = {};
#pragma unroll
  for (int t = 0; t < 8; ++t) {
    int col = n0 + t * 16 + l16;
    float bp = b_proj[col];
#pragma unroll
    for (int j = 0; j < 8; ++j) {
      int row = r0 + half * 8 + j;
      float v = acc[t][j] + bp + x[(size_t)row * DMODEL + col];
      acc[t][j] = v;
      psum[j] += v;
      psq[j]  += v * v;
    }
  }
#pragma unroll
  for (int j = 0; j < 8; ++j) {
    atomicAdd(&Sum[half * 8 + j], psum[j]);
    atomicAdd(&Sq[half * 8 + j],  psq[j]);
  }
  __syncthreads();

  const float inv_d = 1.0f / (float)DMODEL;
#pragma unroll
  for (int t = 0; t < 8; ++t) {
    int col = n0 + t * 16 + l16;
    float g = gamma[col], bb = beta[col];
#pragma unroll
    for (int j = 0; j < 8; ++j) {
      int row = r0 + half * 8 + j;
      float mean = Sum[half * 8 + j] * inv_d;
      float var  = Sq[half * 8 + j] * inv_d - mean * mean;
      float v = (acc[t][j] - mean) * rsqrtf(var + 1e-6f);
      out[(size_t)row * DMODEL + col] = v * g + bb;
    }
  }
}

// ---------------------------------------------------------------------------
// Launch.  Inputs: x, w_q, w_k, w_v, w_proj, b_proj, ln_gamma, ln_beta (f32).
// Workspace layout (bytes):
//   [0,16M)   xb   bf16 [8192,1024]
//   [16M..24M) wqT/wkT/wvT/wpT  bf16 [1024,1024] each (2MB apiece)
//   [24M,40M) Q   [40M,56M) K   [56M,72M) V   [72M,88M) O   (bf16)
// ---------------------------------------------------------------------------
extern "C" void kernel_launch(void* const* d_in, const int* in_sizes, int n_in,
                              void* d_out, int out_size, void* d_ws, size_t ws_size,
                              hipStream_t stream) {
  const float* x      = (const float*)d_in[0];
  const float* w_q    = (const float*)d_in[1];
  const float* w_k    = (const float*)d_in[2];
  const float* w_v    = (const float*)d_in[3];
  const float* w_proj = (const float*)d_in[4];
  const float* b_proj = (const float*)d_in[5];
  const float* gamma  = (const float*)d_in[6];
  const float* beta   = (const float*)d_in[7];
  float* out = (float*)d_out;

  char* ws = (char*)d_ws;
  bf16_t* xb  = (bf16_t*)(ws);
  bf16_t* wqT = (bf16_t*)(ws + (16u << 20));
  bf16_t* wkT = (bf16_t*)(ws + (18u << 20));
  bf16_t* wvT = (bf16_t*)(ws + (20u << 20));
  bf16_t* wpT = (bf16_t*)(ws + (22u << 20));
  bf16_t* Qb  = (bf16_t*)(ws + (24u << 20));
  bf16_t* Kb  = (bf16_t*)(ws + (40u << 20));
  bf16_t* Vb  = (bf16_t*)(ws + (56u << 20));
  bf16_t* Ob  = (bf16_t*)(ws + (72u << 20));

  const int nx = NROWS * DMODEL;  // 8388608
  k_cvt_x<<<nx / (4 * 256), 256, 0, stream>>>(x, xb, nx);

  dim3 gT(DMODEL / 32, DMODEL / 32);
  k_cvt_wT<<<gT, 256, 0, stream>>>(w_q,    wqT);
  k_cvt_wT<<<gT, 256, 0, stream>>>(w_k,    wkT);
  k_cvt_wT<<<gT, 256, 0, stream>>>(w_v,    wvT);
  k_cvt_wT<<<gT, 256, 0, stream>>>(w_proj, wpT);

  k_qkv<<<dim3(DMODEL / 64, NROWS / 128, 3), 128, 0, stream>>>(
      xb, wqT, wkT, wvT, Qb, Kb, Vb);

  k_attn<<<dim3(SEQ / 64, BATCH * NHEADS), 128, 0, stream>>>(Qb, Kb, Vb, Ob);

  k_proj_ln<<<NROWS / 16, 256, 0, stream>>>(Ob, wpT, b_proj, x, gamma, beta, out);
}
